// AuthorGroupAttention_44667659878735
// MI455X (gfx1250) — compile-verified, matrix operand order checked by reading, not run on hardware
//
#include <hip/hip_runtime.h>
#include <hip/hip_bf16.h>

// AuthorGroupAttention on MI455X (gfx1250, wave32, WMMA f16->f32).
//   attn = 0.45*flash(q,k,v) + 0.05*flash(q_r,k_r,v);  out = attn @ Wo^T + bo
// All matmuls via v_wmma_f32_16x16x32_f16.

#define Bn 8
#define Tn 1024
#define En 1024
#define Hn 16
#define Dn 64
#define Gn 4
#define SCALING 0.125f   // D^-0.5

typedef __attribute__((ext_vector_type(16))) _Float16 v16h;
typedef __attribute__((ext_vector_type(8)))  _Float16 h8v;
typedef __attribute__((ext_vector_type(8)))  float    v8f;

union AFrag { v16h v; h8v h[2]; };
union CFrag { v8f  v; float e[8]; };

// Load one 16-bit A/B WMMA fragment. Per ISA layout: lane%16 selects the row
// (M for A, N for B), lane/16 selects the K-half; a lane's 16 halves are the
// two contiguous 8-element chunks at K-offsets {base, base+16}. Both chunks
// are 16B-aligned -> two b128 loads (global or ds).
__device__ __forceinline__ v16h ld_frag(const _Float16* p) {
  AFrag f;
  f.h[0] = *(const h8v*)(p);
  f.h[1] = *(const h8v*)(p + 16);
  return f.v;
}

__device__ __forceinline__ v8f wmma16(v16h a, v16h b, v8f c) {
  return __builtin_amdgcn_wmma_f32_16x16x32_f16(false, a, false, b, (short)0, c,
                                                false, false);
}

__device__ __forceinline__ v8f zero8() {
  v8f z;
#pragma unroll
  for (int i = 0; i < 8; ++i) z[i] = 0.0f;
  return z;
}

// Reductions across a 16-lane group (xor masks 1,2,4,8 keep the two
// half-wave groups independent, matching the C-fragment row mapping).
__device__ __forceinline__ float rmax16(float v) {
#pragma unroll
  for (int m = 1; m < 16; m <<= 1) v = fmaxf(v, __shfl_xor(v, m, 32));
  return v;
}
__device__ __forceinline__ float rsum16(float v) {
#pragma unroll
  for (int m = 1; m < 16; m <<= 1) v += __shfl_xor(v, m, 32);
  return v;
}

__global__ void cvt_f32_to_f16(const float* __restrict__ src,
                               _Float16* __restrict__ dst, int n) {
  int i = blockIdx.x * blockDim.x + threadIdx.x;
  int stride = gridDim.x * blockDim.x;
  for (; i < n; i += stride) dst[i] = (_Float16)src[i];
}

// Generic Y = X @ W^T + bias GEMM (per-batch, optionally routed weights).
// Wave computes a 64x64 tile (4x4 WMMA tiles); WG = 4 waves -> 128x128.
// mode 0: out f32 [B,T,E];  mode 1: out f16 [B,H,T,D];  mode 2: f16 [B,H,D,T].
__global__ __launch_bounds__(128) void gemm_qkv(
    const _Float16* __restrict__ X,    // [B,T,E] f16
    const _Float16* __restrict__ W,    // [E,E] or [G,E,E] f16, rows = N, K-contig
    const float*    __restrict__ bias, // [E] or [G,E]
    const int*      __restrict__ rtok, // nullptr -> shared weights
    float scale, int mode,
    float*    __restrict__ outF,
    _Float16* __restrict__ outH) {
  const int lane = threadIdx.x & 31;
  const int wv   = threadIdx.x >> 5;
  const int b    = blockIdx.z;
  const int Mb   = blockIdx.y * 128 + (wv >> 1) * 64;
  const int Nb   = blockIdx.x * 128 + (wv & 1) * 64;
  const int r    = lane & 15;
  const int hf   = lane >> 4;

  const int g = rtok ? rtok[b] : 0;
  const _Float16* Xb    = X + (size_t)b * Tn * En;
  const _Float16* Wb    = W + (size_t)g * En * En;
  const float*    biasb = bias + (size_t)g * En;

  const _Float16* apt[4];
  const _Float16* bpt[4];
#pragma unroll
  for (int i = 0; i < 4; ++i)
    apt[i] = Xb + (size_t)(Mb + 16 * i + r) * En + hf * 8;
#pragma unroll
  for (int j = 0; j < 4; ++j)
    bpt[j] = Wb + (size_t)(Nb + 16 * j + r) * En + hf * 8;

  v8f C[4][4];
#pragma unroll
  for (int i = 0; i < 4; ++i)
#pragma unroll
    for (int j = 0; j < 4; ++j) C[i][j] = zero8();

  for (int k0 = 0; k0 < En; k0 += 32) {
    v16h A[4], Bf[4];
#pragma unroll
    for (int i = 0; i < 4; ++i) A[i] = ld_frag(apt[i] + k0);
#pragma unroll
    for (int j = 0; j < 4; ++j) Bf[j] = ld_frag(bpt[j] + k0);
#pragma unroll
    for (int i = 0; i < 4; ++i)
#pragma unroll
      for (int j = 0; j < 4; ++j) C[i][j] = wmma16(A[i], Bf[j], C[i][j]);
  }

  // Epilogue. C layout: reg e -> M = e + 8*(lane/16), lane%16 -> N.
#pragma unroll
  for (int j = 0; j < 4; ++j) {
    const int n  = Nb + 16 * j + r;
    const float bn = biasb[n];
#pragma unroll
    for (int i = 0; i < 4; ++i) {
      CFrag cf; cf.v = C[i][j];
#pragma unroll
      for (int e2 = 0; e2 < 8; ++e2) {
        const int t = Mb + 16 * i + e2 + 8 * hf;
        const float val = (cf.e[e2] + bn) * scale;
        if (mode == 0) {
          outF[((size_t)b * Tn + t) * En + n] = val;
        } else {
          const int hd = n >> 6, d = n & 63;
          const size_t base = (size_t)b * Hn + hd;
          if (mode == 1)
            outH[(base * Tn + t) * Dn + d] = (_Float16)val;
          else
            outH[(base * Dn + d) * Tn + t] = (_Float16)val;
        }
      }
    }
  }
}

// Flash attention, one (b,h) per blockIdx.y, 64 queries per WG (16/wave).
// Streams keys in blocks of 64; online softmax; P re-staged via LDS to
// convert C-layout -> A-layout for the P@V WMMAs.
__global__ __launch_bounds__(128) void flash_attn(
    const _Float16* __restrict__ Qm,  // [B,H,T,D] (pre-scaled for q)
    const _Float16* __restrict__ Km,  // [B,H,T,D]
    const _Float16* __restrict__ Vt,  // [B,H,D,T] (transposed V)
    float* __restrict__ attn,         // [B,T,E]
    float wgt, int accum) {
  __shared__ _Float16 lds[4][16][64];
  const int lane = threadIdx.x & 31;
  const int wv   = threadIdx.x >> 5;
  const int bh   = blockIdx.y;
  const int b    = bh >> 4;   // H = 16
  const int hh   = bh & 15;
  const int tq   = blockIdx.x * 64 + wv * 16;
  const int r    = lane & 15;
  const int hf   = lane >> 4;

  const _Float16* Qb = Qm + (size_t)bh * Tn * Dn;
  const _Float16* Kb = Km + (size_t)bh * Tn * Dn;
  const _Float16* Vb = Vt + (size_t)bh * Dn * Tn;

  v16h aq[2];
#pragma unroll
  for (int kk = 0; kk < 2; ++kk)
    aq[kk] = ld_frag(Qb + (size_t)(tq + r) * Dn + kk * 32 + hf * 8);

  CFrag O[4];
#pragma unroll
  for (int dt = 0; dt < 4; ++dt) O[dt].v = zero8();
  float mrow[8], lrow[8];
#pragma unroll
  for (int e2 = 0; e2 < 8; ++e2) { mrow[e2] = -1e30f; lrow[e2] = 0.0f; }

  for (int sb = 0; sb < Tn; sb += 64) {
    if (sb + 64 < Tn) {  // hint next K/V block toward the caches
      __builtin_prefetch(Kb + (size_t)(sb + 64 + r) * Dn, 0, 3);
      __builtin_prefetch(Vb + (size_t)r * Tn + sb + 64, 0, 3);
    }
    // S = Q (16xD) @ K_block^T  -> four 16x16 C tiles
    CFrag S[4];
#pragma unroll
    for (int nt = 0; nt < 4; ++nt) {
      const _Float16* kb = Kb + (size_t)(sb + nt * 16 + r) * Dn + hf * 8;
      v8f s = zero8();
      s = wmma16(aq[0], ld_frag(kb), s);
      s = wmma16(aq[1], ld_frag(kb + 32), s);
      S[nt].v = s;
    }
    // online softmax update (row = e2 + 8*hf lives across the 16-lane group)
    float alpha[8];
#pragma unroll
    for (int e2 = 0; e2 < 8; ++e2) {
      float rm = fmaxf(fmaxf(S[0].e[e2], S[1].e[e2]),
                       fmaxf(S[2].e[e2], S[3].e[e2]));
      rm = rmax16(rm);
      const float mn = fmaxf(mrow[e2], rm);
      alpha[e2] = __expf(mrow[e2] - mn);
      mrow[e2]  = mn;
    }
#pragma unroll
    for (int e2 = 0; e2 < 8; ++e2) {
      float rs = 0.0f;
#pragma unroll
      for (int nt = 0; nt < 4; ++nt) {
        const float p = __expf(S[nt].e[e2] - mrow[e2]);
        S[nt].e[e2] = p;
        rs += p;
      }
      rs = rsum16(rs);
      lrow[e2] = lrow[e2] * alpha[e2] + rs;
#pragma unroll
      for (int dt = 0; dt < 4; ++dt) O[dt].e[e2] *= alpha[e2];
    }
    // stage P (f16) through LDS: C-layout -> A-layout
#pragma unroll
    for (int nt = 0; nt < 4; ++nt)
#pragma unroll
      for (int e2 = 0; e2 < 8; ++e2)
        lds[wv][e2 + 8 * hf][nt * 16 + r] = (_Float16)S[nt].e[e2];
    asm volatile("s_wait_dscnt 0x0" ::: "memory");  // DS in-order, fence compiler
    const v16h ap0 = ld_frag(&lds[wv][r][0] + hf * 8);
    const v16h ap1 = ld_frag(&lds[wv][r][0] + 32 + hf * 8);
    // O += P (16x64) @ V_block (64x64), V^T layout makes B-frags contiguous
#pragma unroll
    for (int dt = 0; dt < 4; ++dt) {
      const _Float16* vb = Vb + (size_t)(dt * 16 + r) * Tn + sb + hf * 8;
      O[dt].v = wmma16(ap0, ld_frag(vb), O[dt].v);
      O[dt].v = wmma16(ap1, ld_frag(vb + 32), O[dt].v);
    }
    asm volatile("s_wait_dscnt 0x0" ::: "memory");
  }

#pragma unroll
  for (int dt = 0; dt < 4; ++dt)
#pragma unroll
    for (int e2 = 0; e2 < 8; ++e2) {
      const int t = tq + e2 + 8 * hf;
      const int o = hh * Dn + dt * 16 + r;
      const size_t idx = ((size_t)b * Tn + t) * En + o;
      const float val = O[dt].e[e2] / lrow[e2] * wgt;
      if (accum) attn[idx] += val; else attn[idx] = val;
    }
}

extern "C" void kernel_launch(void* const* d_in, const int* in_sizes, int n_in,
                              void* d_out, int out_size, void* d_ws, size_t ws_size,
                              hipStream_t stream) {
  const float* hs  = (const float*)d_in[0];
  const int*  rtok = (const int*)d_in[1];
  const float* Wq  = (const float*)d_in[2];
  const float* bq  = (const float*)d_in[3];
  const float* Wk  = (const float*)d_in[4];
  const float* bk  = (const float*)d_in[5];
  const float* Wv  = (const float*)d_in[6];
  const float* bv  = (const float*)d_in[7];
  const float* Wo  = (const float*)d_in[8];
  const float* bo  = (const float*)d_in[9];
  const float* RWq = (const float*)d_in[10];
  const float* Rbq = (const float*)d_in[11];
  const float* RWk = (const float*)d_in[12];
  const float* Rbk = (const float*)d_in[13];
  // d_in[14]/d_in[15] (RWv/Rbv) are dead: reference attends over generic V.

  char* ws = (char*)d_ws;
  size_t off = 0;
  auto alloc = [&](size_t bytes) {
    char* p = ws + off;
    off += (bytes + 255) & ~(size_t)255;
    return (void*)p;
  };
  const size_t BTE = (size_t)Bn * Tn * En;
  _Float16* hs16   = (_Float16*)alloc(BTE * 2);
  _Float16* Wq16   = (_Float16*)alloc((size_t)En * En * 2);
  _Float16* Wk16   = (_Float16*)alloc((size_t)En * En * 2);
  _Float16* Wv16   = (_Float16*)alloc((size_t)En * En * 2);
  _Float16* Wo16   = (_Float16*)alloc((size_t)En * En * 2);
  _Float16* RWq16  = (_Float16*)alloc((size_t)Gn * En * En * 2);
  _Float16* RWk16  = (_Float16*)alloc((size_t)Gn * En * En * 2);
  _Float16* qg     = (_Float16*)alloc(BTE * 2);   // [B,H,T,D]
  _Float16* kg     = (_Float16*)alloc(BTE * 2);   // [B,H,T,D]
  _Float16* vT     = (_Float16*)alloc(BTE * 2);   // [B,H,D,T]
  _Float16* qr     = (_Float16*)alloc(BTE * 2);
  _Float16* kr     = (_Float16*)alloc(BTE * 2);
  float*    attnF  = (float*)   alloc(BTE * 4);   // [B,T,E]
  _Float16* attn16 = (_Float16*)alloc(BTE * 2);

  auto cvt = [&](const float* s, _Float16* d, size_t n) {
    cvt_f32_to_f16<<<4096, 256, 0, stream>>>(s, d, (int)n);
  };
  cvt(hs, hs16, BTE);
  cvt(Wq, Wq16, (size_t)En * En);
  cvt(Wk, Wk16, (size_t)En * En);
  cvt(Wv, Wv16, (size_t)En * En);
  cvt(Wo, Wo16, (size_t)En * En);
  cvt(RWq, RWq16, (size_t)Gn * En * En);
  cvt(RWk, RWk16, (size_t)Gn * En * En);

  dim3 gg(En / 128, Tn / 128, Bn);
  gemm_qkv<<<gg, 128, 0, stream>>>(hs16, Wq16, bq, nullptr, SCALING, 1, nullptr, qg);
  gemm_qkv<<<gg, 128, 0, stream>>>(hs16, Wk16, bk, nullptr, 1.0f,    1, nullptr, kg);
  gemm_qkv<<<gg, 128, 0, stream>>>(hs16, Wv16, bv, nullptr, 1.0f,    2, nullptr, vT);
  gemm_qkv<<<gg, 128, 0, stream>>>(hs16, RWq16, Rbq, rtok, SCALING,  1, nullptr, qr);
  gemm_qkv<<<gg, 128, 0, stream>>>(hs16, RWk16, Rbk, rtok, 1.0f,     1, nullptr, kr);

  dim3 fg(Tn / 64, Bn * Hn);
  flash_attn<<<fg, 128, 0, stream>>>(qg, kg, vT, attnF, 0.45f, 0);
  flash_attn<<<fg, 128, 0, stream>>>(qr, kr, vT, attnF, 0.05f, 1);

  cvt(attnF, attn16, BTE);
  gemm_qkv<<<gg, 128, 0, stream>>>(attn16, Wo16, bo, nullptr, 1.0f, 0,
                                   (float*)d_out, nullptr);
}
